// C_eAR_GenRNN_78503412236558
// MI455X (gfx1250) — matrix operand order repaired and checked by reading, not compiled, not run on hardware
//
#include <hip/hip_runtime.h>
#include <hip/hip_bf16.h>
#include <stdint.h>

// ---------------------------------------------------------------------------
// Model dims
// ---------------------------------------------------------------------------
#define BD    16
#define TD    2048
#define MROWS (BD * TD)      // 32768
#define DEMB  256
#define DCND  640
#define DIN   896            // DEMB + DCND
#define HRNN  896
#define H3    2688           // 3*HRNN
#define HFC   1024
#define NOUT  1024
#define KCH   (HRNN / 32)    // 28 K-chunks in the scan
#define HPAD  (HRNN + 8)     // padded LDS row stride (bank-conflict free)

typedef __attribute__((ext_vector_type(16))) __bf16   v16bf;
typedef __attribute__((ext_vector_type(8)))  __bf16   v8bf;
typedef __attribute__((ext_vector_type(8)))  float    v8f;
typedef __attribute__((ext_vector_type(4)))  unsigned v4u;
typedef __attribute__((ext_vector_type(8)))  int      v8i;
typedef __attribute__((ext_vector_type(4)))  int      v4i;

#if __has_builtin(__builtin_amdgcn_tensor_load_to_lds) && \
    __has_builtin(__builtin_amdgcn_s_wait_tensorcnt)
#define HAVE_TDM 1
#else
#define HAVE_TDM 0
#endif

// f32 -> bf16, round-to-nearest-even
static __device__ inline __bf16 f2bf(float f) {
    unsigned u = __builtin_bit_cast(unsigned, f);
    unsigned r = (u + 0x7FFFu + ((u >> 16) & 1u)) >> 16;
    return __builtin_bit_cast(__bf16, (unsigned short)r);
}

// ---------------------------------------------------------------------------
// WMMA fragment helpers (CDNA5 16x16x32 bf16, wave32; A and B both use contiguous k = 16*half + s)
// ---------------------------------------------------------------------------
static __device__ inline v16bf frag_A(const __bf16* base, int lda, int lane) {
    const int m = lane & 15, half = lane >> 4;
    const __bf16* p = base + (size_t)m * lda + half * 16;
    v8bf lo = *reinterpret_cast<const v8bf*>(p);        // K = half*16 .. +7
    v8bf hi = *reinterpret_cast<const v8bf*>(p + 8);    // K = half*16+8 .. +7
    v16bf a;
#pragma unroll
    for (int j = 0; j < 8; ++j) { a[j] = lo[j]; a[j + 8] = hi[j]; }
    return a;
}

// B[k][n] = W[n][k]; wrow points at W + n0*ldw + k0.
static __device__ inline v16bf frag_B(const __bf16* wrow, int ldw, int lane) {
    const int n = lane & 15, half = lane >> 4;
    const __bf16* p = wrow + (size_t)n * ldw + half * 16;
    return *reinterpret_cast<const v16bf*>(p);          // 32B contiguous
}

static __device__ inline v8f wmma_bf16(v16bf a, v16bf b, v8f c) {
    return __builtin_amdgcn_wmma_f32_16x16x32_bf16(
        false, a, false, b, (short)0, c, false, false);
}

// ---------------------------------------------------------------------------
// Tensor Data Mover: 2D tile load (rows x row_elems bf16) global -> LDS.
// D# bitfields per CDNA5 ISA 8.3/8.4.
// ---------------------------------------------------------------------------
#if HAVE_TDM
static __device__ inline void tdm_load_2d(const void* gaddr, uint32_t lds_off,
                                          uint32_t row_elems, uint32_t rows,
                                          uint32_t stride_elems,
                                          uint32_t dim0, uint32_t dim1) {
    uint64_t ga = (uint64_t)(uintptr_t)gaddr;
    v4u g0;
    g0[0] = 1u;                                            // count=1, user mode
    g0[1] = lds_off;                                       // lds_addr (bytes)
    g0[2] = (uint32_t)ga;                                  // global_addr[31:0]
    g0[3] = (uint32_t)((ga >> 32) & 0x01FFFFFFu) | (2u << 30);  // addr[56:32]|type=2
    v8i g1;
    g1[0] = (int)0x00010000u;                              // data_size=1 (2 bytes)
    g1[1] = (int)((dim0 & 0xFFFFu) << 16);                 // tensor_dim0[15:0]
    g1[2] = (int)((dim0 >> 16) | ((dim1 & 0xFFFFu) << 16));// dim0[31:16]|dim1[15:0]
    g1[3] = (int)((dim1 >> 16) | (row_elems << 16));       // dim1[31:16]|tile_dim0
    g1[4] = (int)rows;                                     // tile_dim1 (tile_dim2=0)
    g1[5] = (int)stride_elems;                             // tensor_dim0_stride[31:0]
    g1[6] = 0;
    g1[7] = 0;
    v4i z4 = {};
#if defined(__clang_major__) && __clang_major__ >= 23
    v8i z8 = {};
    __builtin_amdgcn_tensor_load_to_lds(g0, g1, z4, z4, z8, 0);
#else
    __builtin_amdgcn_tensor_load_to_lds(g0, g1, z4, z4, 0);
#endif
}
#endif

// ---------------------------------------------------------------------------
// Elementwise prep kernels
// ---------------------------------------------------------------------------
__global__ void k_f32_to_bf16(const float* __restrict__ s,
                              __bf16* __restrict__ d, int n) {
    int i = blockIdx.x * blockDim.x + threadIdx.x;
    if (i < n) d[i] = f2bf(s[i]);
}

__global__ void k_embed_concat(const int* __restrict__ sample,
                               const float* __restrict__ cnd,
                               const float* __restrict__ emb,
                               __bf16* __restrict__ xb, int total) {
    int i = blockIdx.x * blockDim.x + threadIdx.x;
    if (i >= total) return;
    int m = i / DIN, c = i - m * DIN;                  // m = b*T + t
    float v;
    if (c < DEMB) { int tok = sample[m]; v = emb[(size_t)tok * DEMB + c]; }
    else          { v = cnd[(size_t)m * DCND + (c - DEMB)]; }
    xb[(size_t)m * DIN + c] = f2bf(v);
}

// ---------------------------------------------------------------------------
// bf16 WMMA GEMM: out[m,n] = sum_k A[m,k]*W[n,k] + bias[n]
// block = 256 thr (8 waves); tile = 32(M) x 128(N); K staged in LDS in
// 128-wide chunks, double buffered; A tile moved by TDM (wave 0).
// MODE 0: f32 store permuted [T][B][H3] (gx)    MODE 1: relu -> bf16 store
// MODE 2: f32 store row-major (final output)
// ---------------------------------------------------------------------------
template <int MODE>
__global__ __launch_bounds__(256) void k_gemm(const __bf16* __restrict__ A,
                                              const __bf16* __restrict__ W,
                                              const float* __restrict__ bias,
                                              void* __restrict__ outp,
                                              int K, int ldo) {
    __shared__ __align__(16) __bf16 As[2][32 * 128];   // 16 KB double buffer
    const int tid  = threadIdx.x;
    const int lane = tid & 31;
    const int wv   = tid >> 5;
    const int m0   = blockIdx.y * 32;
    const int n0   = blockIdx.x * 128 + wv * 16;
    const __bf16* wrow = W + (size_t)n0 * K;
    const int nch = K >> 7;                            // K / 128

#if HAVE_TDM
    const uint32_t lds_off[2] = {
        (uint32_t)(uintptr_t)&As[0][0],                // low 32 bits = LDS offset
        (uint32_t)(uintptr_t)&As[1][0]
    };
#endif

    auto stage = [&](int c) {
#if HAVE_TDM
        if (wv == 0)
            tdm_load_2d(A + (size_t)m0 * K + (size_t)c * 128, lds_off[c & 1],
                        128u, 32u, (uint32_t)K, (uint32_t)K, (uint32_t)MROWS);
#else
        int pos = tid * 8, row = pos >> 7, col = pos & 127;
        *reinterpret_cast<v4u*>(&As[c & 1][row * 128 + col]) =
            *reinterpret_cast<const v4u*>(A + (size_t)(m0 + row) * K +
                                          (size_t)c * 128 + col);
#endif
    };

    v8f acc0 = {}, acc1 = {};
    stage(0);
    for (int c = 0; c < nch; ++c) {
        if (c + 1 < nch) stage(c + 1);
#if HAVE_TDM
        if (wv == 0) {
            if (c + 1 < nch) __builtin_amdgcn_s_wait_tensorcnt(1);
            else             __builtin_amdgcn_s_wait_tensorcnt(0);
        }
#endif
        __syncthreads();                               // chunk c visible in LDS
        const __bf16* ab = &As[c & 1][0];
#pragma unroll
        for (int s = 0; s < 4; ++s) {
            v16bf bf  = frag_B(wrow + (c * 128 + s * 32), K, lane);
            v16bf af0 = frag_A(ab + s * 32, 128, lane);
            v16bf af1 = frag_A(ab + 16 * 128 + s * 32, 128, lane);
            acc0 = wmma_bf16(af0, bf, acc0);
            acc1 = wmma_bf16(af1, bf, acc1);
        }
        __syncthreads();                               // protect buffer reuse
    }

    const int n = n0 + (lane & 15);
    const float bv = bias[n];
#pragma unroll
    for (int h = 0; h < 2; ++h) {
        v8f& acc = h ? acc1 : acc0;
#pragma unroll
        for (int r = 0; r < 8; ++r) {
            int m = m0 + h * 16 + r + ((lane >> 4) << 3);
            float v = acc[r] + bv;
            if (MODE == 0) {
                int b = m >> 11, t = m & (TD - 1);      // m = b*T + t
                ((float*)outp)[((size_t)t * BD + b) * (size_t)ldo + n] = v;
            } else if (MODE == 1) {
                ((__bf16*)outp)[(size_t)m * ldo + n] = f2bf(v > 0.f ? v : 0.f);
            } else {
                ((float*)outp)[(size_t)m * ldo + n] = v;
            }
        }
    }
}

// ---------------------------------------------------------------------------
// GRU scan: 56 persistent workgroups (one per 16 hidden cols), 3 waves each
// (r/z/n gate tiles). W_hh fragments pinned in VGPRs across all 2048 steps.
// h is staged into LDS once per step by all 96 threads (one batched clause,
// 28 KB), so the 28 per-chunk A-fragment reads are short-latency ds_loads
// instead of 28 serialized L2 round trips, and the 3 waves stop loading the
// same data redundantly. LDS rows padded to 904 elems -> conflict-free banks.
// ---------------------------------------------------------------------------
__global__ __launch_bounds__(96) void k_scan(const __bf16* __restrict__ Whh,
                                             const float* __restrict__ gx,
                                             const float* __restrict__ bhh,
                                             __bf16* __restrict__ hb0,
                                             __bf16* __restrict__ hb1,
                                             float* __restrict__ hf0,
                                             float* __restrict__ hf1,
                                             __bf16* __restrict__ hseq,
                                             unsigned* __restrict__ bar) {
    __shared__ __align__(16) __bf16 h_s[16 * HPAD];    // 28.3 KB staged h
    __shared__ float gh_s[3][16][16];
    const int tid  = threadIdx.x;
    const int lane = tid & 31;
    const int g    = tid >> 5;               // gate index: 0=r 1=z 2=n
    const int cb   = blockIdx.x * 16;        // hidden-column base
    const int n0   = cb + g * HRNN;          // column in 3H gate space
    const __bf16* wrow = Whh + (size_t)n0 * HRNN;

    // Pin all W_hh B-fragments for this gate tile in VGPRs (28 x 8 VGPRs).
    v16bf breg[KCH];
#pragma unroll
    for (int kc = 0; kc < KCH; ++kc)
        breg[kc] = frag_B(wrow + kc * 32, HRNN, lane);

    for (int t = 0; t < TD; ++t) {
        const __bf16* hsrc = (t & 1) ? hb1 : hb0;
        __bf16*       hdst = (t & 1) ? hb0 : hb1;
        const float*  hfs  = (t & 1) ? hf1 : hf0;
        float*        hfd  = (t & 1) ? hf0 : hf1;

        // cooperative copy: global h (16x896 bf16) -> LDS, 16B vectors
        for (int v = tid; v < 16 * (HRNN / 8); v += 96) {
            int row = v / (HRNN / 8), col = v - row * (HRNN / 8);
            *reinterpret_cast<v4u*>(&h_s[row * HPAD + col * 8]) =
                *reinterpret_cast<const v4u*>(hsrc + (size_t)row * HRNN + col * 8);
        }
        __syncthreads();

        v8f acc = {};
#pragma unroll
        for (int kc = 0; kc < KCH; ++kc)
            acc = wmma_bf16(frag_A(h_s + kc * 32, HPAD, lane), breg[kc], acc);
#pragma unroll
        for (int r = 0; r < 8; ++r)
            gh_s[g][r + ((lane >> 4) << 3)][lane & 15] = acc[r];
        __syncthreads();

        // gate math for this block's 16 hidden columns (16x16 elements)
        const float* gxt = gx + (size_t)t * (BD * H3);
        for (int e = tid; e < 256; e += 96) {
            int i = e >> 4, jl = e & 15, j = cb + jl;
            float rh = gh_s[0][i][jl] + bhh[j];
            float zh = gh_s[1][i][jl] + bhh[j + HRNN];
            float nh = gh_s[2][i][jl] + bhh[j + 2 * HRNN];
            const float* gxr = gxt + (size_t)i * H3;
            float rx = gxr[j], zx = gxr[j + HRNN], nx = gxr[j + 2 * HRNN];
            float r  = 1.f / (1.f + __expf(-(rx + rh)));
            float z  = 1.f / (1.f + __expf(-(zx + zh)));
            float nn = tanhf(nx + r * nh);
            float hp = hfs[(size_t)i * HRNN + j];
            float hn = (1.f - z) * nn + z * hp;
            hfd[(size_t)i * HRNN + j]  = hn;
            hdst[(size_t)i * HRNN + j] = f2bf(hn);
            hseq[((size_t)i * TD + t) * HRNN + j] = f2bf(hn);   // row = b*T+t
        }
        __threadfence();
        __syncthreads();
        if (tid == 0) {
            atomicAdd(bar, 1u);
            unsigned tgt = (unsigned)(t + 1) * gridDim.x;
            while (__hip_atomic_load(bar, __ATOMIC_RELAXED,
                                     __HIP_MEMORY_SCOPE_AGENT) < tgt)
                __builtin_amdgcn_s_sleep(8);
        }
        __syncthreads();
    }
}

// ---------------------------------------------------------------------------
// Host launch
// ---------------------------------------------------------------------------
extern "C" void kernel_launch(void* const* d_in, const int* in_sizes, int n_in,
                              void* d_out, int out_size, void* d_ws, size_t ws_size,
                              hipStream_t stream) {
    const int*   sample = (const int*)d_in[0];
    const float* cnd    = (const float*)d_in[1];
    const float* emb    = (const float*)d_in[2];
    const float* W_ih   = (const float*)d_in[3];
    const float* b_ih   = (const float*)d_in[4];
    const float* W_hh   = (const float*)d_in[5];
    const float* b_hh   = (const float*)d_in[6];
    const float* fc1_w  = (const float*)d_in[7];
    const float* fc1_b  = (const float*)d_in[8];
    const float* fc2_w  = (const float*)d_in[9];
    const float* fc2_b  = (const float*)d_in[10];
    float* out = (float*)d_out;

    // workspace carve-out (256B aligned); total ~551 MB
    char* w = (char*)d_ws;
    size_t off = 0;
    auto carve = [&](size_t bytes) -> char* {
        char* p = w + off;
        off = (off + bytes + 255) & ~(size_t)255;
        return p;
    };
    __bf16* xb    = (__bf16*)carve((size_t)MROWS * DIN * 2);
    __bf16* Wih_b = (__bf16*)carve((size_t)H3 * DIN * 2);
    __bf16* Whh_b = (__bf16*)carve((size_t)H3 * HRNN * 2);
    __bf16* f1w_b = (__bf16*)carve((size_t)HFC * HRNN * 2);
    __bf16* f2w_b = (__bf16*)carve((size_t)NOUT * HFC * 2);
    float*  gxp   = (float*)carve((size_t)TD * BD * H3 * 4);
    __bf16* hb0   = (__bf16*)carve((size_t)BD * HRNN * 2);
    __bf16* hb1   = (__bf16*)carve((size_t)BD * HRNN * 2);
    float*  hf0   = (float*)carve((size_t)BD * HRNN * 4);
    float*  hf1   = (float*)carve((size_t)BD * HRNN * 4);
    __bf16* hseq  = (__bf16*)carve((size_t)MROWS * HRNN * 2);
    __bf16* a1    = (__bf16*)carve((size_t)MROWS * HFC * 2);
    unsigned* bar = (unsigned*)carve(256);

    const int thr = 256;
    k_f32_to_bf16<<<(H3 * DIN + thr - 1) / thr, thr, 0, stream>>>(W_ih, Wih_b, H3 * DIN);
    k_f32_to_bf16<<<(H3 * HRNN + thr - 1) / thr, thr, 0, stream>>>(W_hh, Whh_b, H3 * HRNN);
    k_f32_to_bf16<<<(HFC * HRNN + thr - 1) / thr, thr, 0, stream>>>(fc1_w, f1w_b, HFC * HRNN);
    k_f32_to_bf16<<<(NOUT * HFC + thr - 1) / thr, thr, 0, stream>>>(fc2_w, f2w_b, NOUT * HFC);
    k_embed_concat<<<((size_t)MROWS * DIN + thr - 1) / thr, thr, 0, stream>>>(
        sample, cnd, emb, xb, MROWS * DIN);

    // gx = x @ W_ih^T + b_ih, stored [T][B][3H] f32
    {
        dim3 g(H3 / 128, MROWS / 32);
        k_gemm<0><<<g, 256, 0, stream>>>(xb, Wih_b, b_ih, gxp, DIN, H3);
    }

    (void)hipMemsetAsync(hb0, 0, (size_t)BD * HRNN * 2, stream);
    (void)hipMemsetAsync(hb1, 0, (size_t)BD * HRNN * 2, stream);
    (void)hipMemsetAsync(hf0, 0, (size_t)BD * HRNN * 4, stream);
    (void)hipMemsetAsync(hf1, 0, (size_t)BD * HRNN * 4, stream);
    (void)hipMemsetAsync(bar, 0, 256, stream);

    k_scan<<<HRNN / 16, 96, 0, stream>>>(Whh_b, gxp, b_hh, hb0, hb1, hf0, hf1,
                                         hseq, bar);

    {
        dim3 g(HFC / 128, MROWS / 32);
        k_gemm<1><<<g, 256, 0, stream>>>(hseq, f1w_b, fc1_b, a1, HRNN, HFC);
    }
    {
        dim3 g(NOUT / 128, MROWS / 32);
        k_gemm<2><<<g, 256, 0, stream>>>(a1, f2w_b, fc2_b, out, HFC, NOUT);
    }
    (void)in_sizes; (void)n_in; (void)out_size; (void)ws_size;
}
